// LlamaAttention_hqe_10763188044286
// MI455X (gfx1250) — matrix operand order repaired and checked by reading, not compiled, not run on hardware
//
#include <hip/hip_runtime.h>
#include <hip/hip_bf16.h>
#include <cstdint>

// ---------------------------------------------------------------------------
// Problem constants (from reference)
// ---------------------------------------------------------------------------
#define SEQ  2048
#define DMOD 4096
#define NH   32
#define NKVH 8
#define DH   128
#define NREP (NH / NKVH)

typedef __attribute__((ext_vector_type(16))) _Float16 v16h;
typedef __attribute__((ext_vector_type(8)))  _Float16 v8h;
typedef __attribute__((ext_vector_type(8)))  float    v8f;

// ---------------------------------------------------------------------------
// WMMA helpers (CDNA5 wave32, 16x16x32 f16 -> f32)
// ---------------------------------------------------------------------------
__device__ inline v16h join8(v8h lo, v8h hi) {
    return __builtin_shufflevector(lo, hi, 0,1,2,3,4,5,6,7,8,9,10,11,12,13,14,15);
}

__device__ inline v8f wmma_f16(v16h a, v16h b, v8f c) {
    // (neg_a, A, neg_b, B, c_mod, C, reuse_a, reuse_b)
    return __builtin_amdgcn_wmma_f32_16x16x32_f16(false, a, false, b, (short)0, c, false, false);
}

// A-fragment (16x32 f16, MxK): lane l -> row l%16; VGPR0-3 hold K=[8*half..+7],
// VGPR4-7 hold K=[16+8*half..+7]. Source is row-major [rows][ld], K contiguous.
__device__ inline v16h load_frag_a(const _Float16* base, int ld) {
    int lane = threadIdx.x & 31;
    const _Float16* p = base + (size_t)(lane & 15) * ld + ((lane >> 4) << 3);
    v8h lo = *(const v8h*)(p);
    v8h hi = *(const v8h*)(p + 16);
    return join8(lo, hi);
}

// B-fragment (32x16 f16, KxN): lane l -> col l%16; VGPR0-3 hold K=[8*half..+7],
// VGPR4-7 hold K=[16+8*half..+7].
// Source is the TRANSPOSED matrix, row-major [N][ld], K contiguous per row.
__device__ inline v16h load_frag_b(const _Float16* baseT, int ld) {
    int lane = threadIdx.x & 31;
    const _Float16* p = baseT + (size_t)(lane & 15) * ld + ((lane >> 4) << 3);
    return join8(*(const v8h*)(p), *(const v8h*)(p + 16));
}

// C/D 16x16 f32 tile: VGPR i -> row i + 8*(lane/16), col lane%16
__device__ inline void store_tile_f32(float* C, int ldc, v8f c) {
    int lane = threadIdx.x & 31;
    int col = lane & 15;
    int rb  = (lane >> 4) << 3;
#pragma unroll
    for (int i = 0; i < 8; ++i)
        C[(size_t)(rb + i) * ldc + col] = c[i];
}

// ---------------------------------------------------------------------------
// Elementwise conversion kernels
// ---------------------------------------------------------------------------
__global__ void cast_f32_to_f16(const float* __restrict__ in,
                                _Float16* __restrict__ out, size_t n) {
    size_t i = (size_t)blockIdx.x * blockDim.x + threadIdx.x;
    if (i < n) out[i] = (_Float16)in[i];
}

// W [K][N] (f32, row-major) -> WT [N][K] (f16, row-major)
__global__ void transpose_cast_f16(const float* __restrict__ W,
                                   _Float16* __restrict__ WT, int K, int N) {
    size_t i = (size_t)blockIdx.x * blockDim.x + threadIdx.x;
    if (i < (size_t)K * N) {
        int k = (int)(i / N);
        int n = (int)(i % N);
        WT[(size_t)n * K + k] = (_Float16)W[i];
    }
}

// RoPE + head-major relayout:
// X [S][nh*DH] f32  ->  Xh [nh][S][DH] f16, with llama rotate_half rotation
__global__ void rope_relayout(const float* __restrict__ X,
                              const float* __restrict__ cosT,
                              const float* __restrict__ sinT,
                              _Float16* __restrict__ Xh, int nh) {
    size_t i = (size_t)blockIdx.x * blockDim.x + threadIdx.x;
    size_t total = (size_t)SEQ * nh * DH;
    if (i >= total) return;
    int d = (int)(i % DH);
    int h = (int)((i / DH) % nh);
    int s = (int)(i / ((size_t)DH * nh));
    float x  = X[i];
    float xr = (d < DH / 2) ? -X[i + DH / 2] : X[i - DH / 2];
    float c  = cosT[(size_t)s * DH + d];
    float sn = sinT[(size_t)s * DH + d];
    Xh[((size_t)h * SEQ + s) * DH + d] = (_Float16)(x * c + xr * sn);
}

// V [S][NKVH*DH] f32 -> Vt [NKVH][DH][S] f16 (transposed: dh-major rows)
__global__ void v_transpose_cast(const float* __restrict__ V,
                                 _Float16* __restrict__ Vt) {
    size_t i = (size_t)blockIdx.x * blockDim.x + threadIdx.x;
    size_t total = (size_t)SEQ * NKVH * DH;
    if (i >= total) return;
    int d = (int)(i % DH);
    int h = (int)((i / DH) % NKVH);
    int s = (int)(i / ((size_t)DH * NKVH));
    Vt[((size_t)h * DH + d) * SEQ + s] = (_Float16)V[i];
}

// ---------------------------------------------------------------------------
// WMMA GEMM (NT): C[M][N] f32 = A[M][K] f16  x  BT[N][K] f16
// Block: 256 threads = 8 waves (2x4), block tile 128(M) x 256(N).
// Wave tile 64x64 = 4x4 WMMA tiles -> 16 b128 loads per 16 WMMAs (1 load/WMMA).
// ---------------------------------------------------------------------------
__global__ void __launch_bounds__(256)
gemm_nt_wmma(const _Float16* __restrict__ A, const _Float16* __restrict__ BT,
             float* __restrict__ C, int M, int N, int K) {
    int wave = threadIdx.x >> 5;
    int wm = wave >> 2;                 // 0..1
    int wn = wave & 3;                  // 0..3
    int m0 = blockIdx.y * 128 + wm * 64;
    int n0 = blockIdx.x * 256 + wn * 64;
    if (m0 >= M || n0 >= N) return;

    v8f acc[4][4];
#pragma unroll
    for (int i = 0; i < 4; ++i)
#pragma unroll
        for (int j = 0; j < 4; ++j) acc[i][j] = (v8f){0,0,0,0,0,0,0,0};

    const _Float16* Abase = A + (size_t)m0 * K;
    const _Float16* Bbase = BT + (size_t)n0 * K;

    for (int k0 = 0; k0 < K; k0 += 32) {
        v16h af[4], bf[4];
#pragma unroll
        for (int i = 0; i < 4; ++i)
            af[i] = load_frag_a(Abase + (size_t)(i * 16) * K + k0, K);
#pragma unroll
        for (int j = 0; j < 4; ++j)
            bf[j] = load_frag_b(Bbase + (size_t)(j * 16) * K + k0, K);
#pragma unroll
        for (int i = 0; i < 4; ++i)
#pragma unroll
            for (int j = 0; j < 4; ++j)
                acc[i][j] = wmma_f16(af[i], bf[j], acc[i][j]);
    }

#pragma unroll
    for (int i = 0; i < 4; ++i)
#pragma unroll
        for (int j = 0; j < 4; ++j)
            store_tile_f32(C + (size_t)(m0 + i * 16) * N + n0 + j * 16, N, acc[i][j]);
}

// ---------------------------------------------------------------------------
// Flash attention with WMMA.
// Grid: (SEQ/64, NH). Block: 128 threads = 4 waves; each wave owns 16 queries.
// Qh [NH][S][DH] f16 (RoPE applied), Kh [NKVH][S][DH] f16, Vt [NKVH][DH][S] f16
// Output: attn [S][NH*DH] f16 (row-major, ready to be A of the Wo GEMM)
// ---------------------------------------------------------------------------
__global__ void __launch_bounds__(128)
flash_attn_wmma(const _Float16* __restrict__ Qh, const _Float16* __restrict__ Kh,
                const _Float16* __restrict__ Vt, _Float16* __restrict__ attnOut) {
    __shared__ __align__(32) _Float16 pbuf[4][16 * 32];   // per-wave P tile

    int wave = threadIdx.x >> 5;
    int lane = threadIdx.x & 31;
    int col  = lane & 15;
    int rb   = (lane >> 4) << 3;

    int h   = blockIdx.y;
    int kvh = h / NREP;
    int q0  = blockIdx.x * 64 + wave * 16;

    const _Float16* Qbase = Qh + ((size_t)h * SEQ + q0) * DH;
    const _Float16* Kbase = Kh + (size_t)kvh * SEQ * DH;
    const _Float16* Vbase = Vt + (size_t)kvh * DH * SEQ;

    // Q fragments for this wave's 16 rows (K = DH = 128 -> 4 fragments)
    v16h qf0 = load_frag_a(Qbase + 0,  DH);
    v16h qf1 = load_frag_a(Qbase + 32, DH);
    v16h qf2 = load_frag_a(Qbase + 64, DH);
    v16h qf3 = load_frag_a(Qbase + 96, DH);

    v8f o[8];
#pragma unroll
    for (int t = 0; t < 8; ++t) o[t] = (v8f){0,0,0,0,0,0,0,0};
    float m[8], l[8];
#pragma unroll
    for (int i = 0; i < 8; ++i) { m[i] = -1.0e30f; l[i] = 0.0f; }

    const float scale = 0.08838834764831845f;  // 1/sqrt(128)

    for (int j0 = 0; j0 < q0 + 16; j0 += 32) {
        // ---- scores: two 16x16 tiles (keys j0..+15, j0+16..+31) ----
        v8f s0 = {0,0,0,0,0,0,0,0};
        v8f s1 = {0,0,0,0,0,0,0,0};
#pragma unroll
        for (int kf = 0; kf < 4; ++kf) {
            v16h a = (kf == 0) ? qf0 : (kf == 1) ? qf1 : (kf == 2) ? qf2 : qf3;
            v16h b0 = load_frag_b(Kbase + (size_t)j0 * DH + kf * 32, DH);
            v16h b1 = load_frag_b(Kbase + (size_t)(j0 + 16) * DH + kf * 32, DH);
            s0 = wmma_f16(a, b0, s0);
            s1 = wmma_f16(a, b1, s1);
        }
        // ---- scale + causal mask (C layout: row = q0+i+rb, col = j0+t*16+col) ----
#pragma unroll
        for (int i = 0; i < 8; ++i) {
            int row = q0 + rb + i;
            float v0 = s0[i] * scale;
            float v1 = s1[i] * scale;
            if (j0 + col > row)      v0 = -3.0e38f;
            if (j0 + 16 + col > row) v1 = -3.0e38f;
            s0[i] = v0;
            s1[i] = v1;
        }
        // ---- row max over this key block (xor-reduce within 16-lane halves) ----
        float mn[8], sc[8];
#pragma unroll
        for (int i = 0; i < 8; ++i) {
            float bm = fmaxf(s0[i], s1[i]);
            bm = fmaxf(bm, __shfl_xor(bm, 1, 32));
            bm = fmaxf(bm, __shfl_xor(bm, 2, 32));
            bm = fmaxf(bm, __shfl_xor(bm, 4, 32));
            bm = fmaxf(bm, __shfl_xor(bm, 8, 32));
            mn[i] = fmaxf(m[i], bm);
            sc[i] = __expf(m[i] - mn[i]);
            m[i]  = mn[i];
        }
        // ---- exponentiate + row sum ----
#pragma unroll
        for (int i = 0; i < 8; ++i) {
            s0[i] = __expf(s0[i] - mn[i]);
            s1[i] = __expf(s1[i] - mn[i]);
            float rs = s0[i] + s1[i];
            rs += __shfl_xor(rs, 1, 32);
            rs += __shfl_xor(rs, 2, 32);
            rs += __shfl_xor(rs, 4, 32);
            rs += __shfl_xor(rs, 8, 32);
            l[i] = l[i] * sc[i] + rs;
        }
        // ---- rescale O accumulators ----
#pragma unroll
        for (int t = 0; t < 8; ++t)
#pragma unroll
            for (int i = 0; i < 8; ++i) o[t][i] *= sc[i];

        // ---- P (C layout) -> LDS -> A-fragment layout ----
        _Float16* pb = pbuf[wave];
#pragma unroll
        for (int i = 0; i < 8; ++i) {
            pb[(rb + i) * 32 + col]      = (_Float16)s0[i];
            pb[(rb + i) * 32 + 16 + col] = (_Float16)s1[i];
        }
        v16h pf = load_frag_a(pb, 32);   // DS ops are in-order within a wave

        // ---- O += P x V  (8 dh-tiles, contraction over 32 keys) ----
#pragma unroll
        for (int dt = 0; dt < 8; ++dt) {
            v16h vf = load_frag_b(Vbase + (size_t)(dt * 16) * SEQ + j0, SEQ);
            o[dt] = wmma_f16(pf, vf, o[dt]);
        }
    }

    // ---- epilogue: O / l -> attn [S][NH*DH] f16 ----
#pragma unroll
    for (int dt = 0; dt < 8; ++dt) {
#pragma unroll
        for (int i = 0; i < 8; ++i) {
            float val = o[dt][i] / l[i];
            attnOut[(size_t)(q0 + rb + i) * (NH * DH) + h * DH + dt * 16 + col] =
                (_Float16)val;
        }
    }
}

// ---------------------------------------------------------------------------
// Host launcher
// ---------------------------------------------------------------------------
extern "C" void kernel_launch(void* const* d_in, const int* in_sizes, int n_in,
                              void* d_out, int out_size, void* d_ws, size_t ws_size,
                              hipStream_t stream) {
    (void)in_sizes; (void)n_in; (void)out_size; (void)ws_size;

    const float* hidden = (const float*)d_in[0];   // [S][D]
    const float* cosT   = (const float*)d_in[1];   // [S][DH]
    const float* sinT   = (const float*)d_in[2];   // [S][DH]
    // d_in[3] = attention_mask (pure causal -> implemented analytically)
    const float* Wq     = (const float*)d_in[4];   // [D][NH*DH]
    const float* Wk     = (const float*)d_in[5];   // [D][NKVH*DH]
    const float* Wv     = (const float*)d_in[6];   // [D][NKVH*DH]
    const float* Wo     = (const float*)d_in[7];   // [NH*DH][D]
    float* out          = (float*)d_out;           // [S][D]

    char* w = (char*)d_ws;
    size_t off = 0;
    auto walloc = [&](size_t bytes) -> void* {
        void* p = (void*)(w + off);
        off += (bytes + 255) & ~(size_t)255;
        return p;
    };

    _Float16* Ah   = (_Float16*)walloc((size_t)SEQ * DMOD * 2);
    _Float16* WqT  = (_Float16*)walloc((size_t)DMOD * (NH * DH) * 2);
    _Float16* WkT  = (_Float16*)walloc((size_t)DMOD * (NKVH * DH) * 2);
    _Float16* WvT  = (_Float16*)walloc((size_t)DMOD * (NKVH * DH) * 2);
    _Float16* WoT  = (_Float16*)walloc((size_t)DMOD * DMOD * 2);
    float*    Qf   = (float*)walloc((size_t)SEQ * NH * DH * 4);
    float*    Kf   = (float*)walloc((size_t)SEQ * NKVH * DH * 4);
    float*    Vf   = (float*)walloc((size_t)SEQ * NKVH * DH * 4);
    _Float16* Qh   = (_Float16*)walloc((size_t)SEQ * NH * DH * 2);
    _Float16* Kh2  = (_Float16*)walloc((size_t)SEQ * NKVH * DH * 2);
    _Float16* Vt   = (_Float16*)walloc((size_t)SEQ * NKVH * DH * 2);
    _Float16* attn = (_Float16*)walloc((size_t)SEQ * NH * DH * 2);

    const int TB = 256;
    auto blocks = [&](size_t n) { return (unsigned)((n + TB - 1) / TB); };

    // 1) casts / transposes to WMMA-friendly f16 layouts
    cast_f32_to_f16<<<blocks((size_t)SEQ * DMOD), TB, 0, stream>>>(
        hidden, Ah, (size_t)SEQ * DMOD);
    transpose_cast_f16<<<blocks((size_t)DMOD * NH * DH), TB, 0, stream>>>(
        Wq, WqT, DMOD, NH * DH);
    transpose_cast_f16<<<blocks((size_t)DMOD * NKVH * DH), TB, 0, stream>>>(
        Wk, WkT, DMOD, NKVH * DH);
    transpose_cast_f16<<<blocks((size_t)DMOD * NKVH * DH), TB, 0, stream>>>(
        Wv, WvT, DMOD, NKVH * DH);
    transpose_cast_f16<<<blocks((size_t)DMOD * DMOD), TB, 0, stream>>>(
        Wo, WoT, NH * DH, DMOD);

    // 2) QKV projections (WMMA GEMMs); block tile 128x256
    gemm_nt_wmma<<<dim3((NH * DH) / 256, SEQ / 128), 256, 0, stream>>>(
        Ah, WqT, Qf, SEQ, NH * DH, DMOD);
    gemm_nt_wmma<<<dim3((NKVH * DH) / 256, SEQ / 128), 256, 0, stream>>>(
        Ah, WkT, Kf, SEQ, NKVH * DH, DMOD);
    gemm_nt_wmma<<<dim3((NKVH * DH) / 256, SEQ / 128), 256, 0, stream>>>(
        Ah, WvT, Vf, SEQ, NKVH * DH, DMOD);

    // 3) RoPE + relayout to attention-operand layouts
    rope_relayout<<<blocks((size_t)SEQ * NH * DH), TB, 0, stream>>>(
        Qf, cosT, sinT, Qh, NH);
    rope_relayout<<<blocks((size_t)SEQ * NKVH * DH), TB, 0, stream>>>(
        Kf, cosT, sinT, Kh2, NKVH);
    v_transpose_cast<<<blocks((size_t)SEQ * NKVH * DH), TB, 0, stream>>>(Vf, Vt);

    // 4) flash attention (causal, online softmax, GQA via kvh = h/4)
    flash_attn_wmma<<<dim3(SEQ / 64, NH), 128, 0, stream>>>(Qh, Kh2, Vt, attn);

    // 5) output projection -> f32 d_out
    gemm_nt_wmma<<<dim3(DMOD / 256, SEQ / 128), 256, 0, stream>>>(
        attn, WoT, out, SEQ, DMOD, DMOD);
}
